// GGNN_VCG_42047729827849
// MI455X (gfx1250) — matrix operand order repaired and checked by reading, not compiled, not run on hardware
//
#include <hip/hip_runtime.h>

#define DIM 128

typedef unsigned short u16;
typedef __bf16 bf16_t;
typedef bf16_t v16bf __attribute__((ext_vector_type(16)));
typedef bf16_t v8bf  __attribute__((ext_vector_type(8)));
typedef float  v8f   __attribute__((ext_vector_type(8)));

// ---------------- helpers ----------------

__device__ __forceinline__ u16 f2bf(float f) {
  unsigned u = __float_as_uint(f);
  u += 0x7FFFu + ((u >> 16) & 1u);   // round-to-nearest-even
  return (u16)(u >> 16);
}

__device__ __forceinline__ unsigned long long pack4(float4 v) {
  return (unsigned long long)f2bf(v.x)
       | ((unsigned long long)f2bf(v.y) << 16)
       | ((unsigned long long)f2bf(v.z) << 32)
       | ((unsigned long long)f2bf(v.w) << 48);
}

// Load WMMA A-operand (16x32 bf16 tile, K-step ks) from a row-major [16 x ld]
// bf16 buffer. Lane l holds row M=l%16; element e covers
// K=kb+e (contiguous, matching load_bT) with kb = (l/16)*16.
__device__ __forceinline__ v16bf load_a(const u16* base, int l16, int hf, int ks, int ld) {
  const int kb = ks * 32 + hf * 16;
  const v8bf lo = *(const v8bf*)(base + l16 * ld + kb);
  const v8bf hi = *(const v8bf*)(base + l16 * ld + kb + 8);
  v16bf a;
#pragma unroll
  for (int i = 0; i < 8; ++i) { a[i] = lo[i]; a[i + 8] = hi[i]; }
  return a;
}

// B-operand for Y = X @ W^T : B[k, n] = W[n, k]. In this code's contiguous k order, lane l
// element e holds (K = (l/16)*16 + e, N = l%16) -> 32 contiguous bytes of W's
// row (jt*16 + l%16). ld = K dimension of W's rows.
__device__ __forceinline__ v16bf load_bT(const u16* W, int jt, int l16, int hf, int ks, int ld) {
  return *(const v16bf*)(W + (size_t)(jt * 16 + l16) * ld + ks * 32 + hf * 16);
}

__device__ __forceinline__ v8f wmma_bf16(v16bf a, v16bf b, v8f c) {
  return __builtin_amdgcn_wmma_f32_16x16x32_bf16(false, a, false, b, (short)0, c, false, false);
}

__device__ __forceinline__ float sigmoidf_(float x) { return 1.f / (1.f + __expf(-x)); }
__device__ __forceinline__ float tanhf_(float x)    { return 2.f / (1.f + __expf(-2.f * x)) - 1.f; }

// ---------------- utility kernels ----------------

__global__ void k_cast(const float* __restrict__ s, u16* __restrict__ d, int n) {
  int i = blockIdx.x * blockDim.x + threadIdx.x;
  if (i < n) d[i] = f2bf(s[i]);
}

__global__ void k_copy4(const float4* __restrict__ s, float4* __restrict__ d, int n4) {
  int i = blockIdx.x * blockDim.x + threadIdx.x;
  if (i < n4) d[i] = s[i];
}

__global__ void k_zero4(float4* __restrict__ d, int n4) {
  int i = blockIdx.x * blockDim.x + threadIdx.x;
  if (i < n4) d[i] = make_float4(0.f, 0.f, 0.f, 0.f);
}

// ---------------- fused 2-layer MLP (WMMA bf16) ----------------
// Y = relu(X @ W1^T + b1) @ W2^T + b2 ; X,Y fp32 [N,128]; W1b/W2b bf16 [128,128]
__global__ __launch_bounds__(256) void k_mlp(
    const float* __restrict__ X,
    const u16* __restrict__ W1b, const float* __restrict__ b1,
    const u16* __restrict__ W2b, const float* __restrict__ b2,
    float* __restrict__ Y, int N)
{
  __shared__ u16 lds[8 * 16 * DIM];             // 32 KB: one 16x128 bf16 tile per wave
  const int tid  = threadIdx.x;
  const int wave = tid >> 5, lane = tid & 31;
  const int hf   = lane >> 4, l16 = lane & 15;
  const int row0 = (blockIdx.x * 8 + wave) * 16;
  if (row0 >= N) return;                        // uniform per wave; no barriers used
  u16* tile = lds + wave * 16 * DIM;

  // stage X tile fp32 -> bf16 (row-major 16x128), rows clamped for tail
#pragma unroll
  for (int r = 0; r < 16; ++r) {
    int rr = row0 + r; if (rr > N - 1) rr = N - 1;
    const float4 v = *(const float4*)(X + (size_t)rr * DIM + lane * 4);
    *(unsigned long long*)(tile + r * DIM + lane * 4) = pack4(v);
  }

  v16bf A[4];
  v8f acc[8];

  // ---- layer 1: H = relu(X W1^T + b1) ----
#pragma unroll
  for (int ks = 0; ks < 4; ++ks) A[ks] = load_a(tile, l16, hf, ks, DIM);
#pragma unroll
  for (int jt = 0; jt < 8; ++jt) {
    v8f c = {0.f, 0.f, 0.f, 0.f, 0.f, 0.f, 0.f, 0.f};
#pragma unroll
    for (int ks = 0; ks < 4; ++ks)
      c = wmma_bf16(A[ks], load_bT(W1b, jt, l16, hf, ks, DIM), c);
    acc[jt] = c;
  }
  // bias + relu; write H back into the same LDS tile as bf16 (X fully consumed)
#pragma unroll
  for (int jt = 0; jt < 8; ++jt) {
    const int col = jt * 16 + l16;
    const float bb = b1[col];
#pragma unroll
    for (int v = 0; v < 8; ++v) {
      float val = acc[jt][v] + bb;
      val = val > 0.f ? val : 0.f;
      tile[(v + 8 * hf) * DIM + col] = f2bf(val);
    }
  }

  // ---- layer 2: Y = H W2^T + b2 ----
#pragma unroll
  for (int ks = 0; ks < 4; ++ks) A[ks] = load_a(tile, l16, hf, ks, DIM);
#pragma unroll
  for (int jt = 0; jt < 8; ++jt) {
    v8f c = {0.f, 0.f, 0.f, 0.f, 0.f, 0.f, 0.f, 0.f};
#pragma unroll
    for (int ks = 0; ks < 4; ++ks)
      c = wmma_bf16(A[ks], load_bT(W2b, jt, l16, hf, ks, DIM), c);
    const int col = jt * 16 + l16;
    const float bb = b2[col];
#pragma unroll
    for (int v = 0; v < 8; ++v) {
      const int gr = row0 + v + 8 * hf;
      if (gr < N) Y[(size_t)gr * DIM + col] = c[v] + bb;
    }
  }
}

// ---------------- edge gather + segment-sum scatter ----------------
// aggr[dmap[eidx[i]]] += M[smap[eidx[i]]]   (128 floats per edge, 4/lane)
__global__ __launch_bounds__(256) void k_scatter(
    const float* __restrict__ M, const int* __restrict__ eidx,
    const int* __restrict__ smap, const int* __restrict__ dmap,
    float* __restrict__ aggr, int EP)
{
  const int gid = blockIdx.x * 256 + threadIdx.x;
  const int i = gid >> 5;
  if (i >= EP) return;
  const int lane = threadIdx.x & 31;
  const int e = eidx[i];
  const int s = smap[e];
  const int d = dmap[e];
  const float4 v = *(const float4*)(M + (size_t)s * DIM + lane * 4);
  float* dst = aggr + (size_t)d * DIM + lane * 4;
  unsafeAtomicAdd(dst + 0, v.x);
  unsafeAtomicAdd(dst + 1, v.y);
  unsafeAtomicAdd(dst + 2, v.z);
  unsafeAtomicAdd(dst + 3, v.w);
}

// ---------------- GRU cell (WMMA bf16 + fused elementwise) ----------------
// x = concat(xa, xb) [N,256]; gi = x@Wih^T + bih; gh = h@Whh^T + bhh (384 wide)
// hout = (1-z)*tanh(ig + r*hg) + z*h
__global__ __launch_bounds__(256) void k_gru(
    const float* __restrict__ xa, const float* __restrict__ xb,
    const float* __restrict__ h,
    const u16* __restrict__ Wihb, const u16* __restrict__ Whhb,
    const float* __restrict__ bih, const float* __restrict__ bhh,
    float* __restrict__ hout, int N)
{
  __shared__ u16 xt[16 * 256];   // 8 KB
  __shared__ u16 ht[16 * 128];   // 4 KB
  const int tid  = threadIdx.x;
  const int row0 = blockIdx.x * 16;

  {                               // cooperative stage: fp32 -> bf16
    const int r = tid >> 4;
    int rr = row0 + r; if (rr > N - 1) rr = N - 1;
    const int cseg = (tid & 15) * 16;
    const float* src = (cseg < 128) ? (xa + (size_t)rr * 128 + cseg)
                                    : (xb + (size_t)rr * 128 + (cseg - 128));
#pragma unroll
    for (int q = 0; q < 4; ++q) {
      const float4 v = *(const float4*)(src + q * 4);
      *(unsigned long long*)(xt + r * 256 + cseg + q * 4) = pack4(v);
    }
    const int c8 = (tid & 15) * 8;
#pragma unroll
    for (int q = 0; q < 2; ++q) {
      const float4 v = *(const float4*)(h + (size_t)rr * 128 + c8 + q * 4);
      *(unsigned long long*)(ht + r * 128 + c8 + q * 4) = pack4(v);
    }
  }
  __syncthreads();

  const int wave = tid >> 5, lane = tid & 31;
  const int hf = lane >> 4, l16 = lane & 15;

  v8f ai[3], ah[3];
#pragma unroll
  for (int g = 0; g < 3; ++g) {
    const int jt = wave + g * 8;                 // wave w owns cols 16w of r/z/n slabs
    v8f ci = {0.f, 0.f, 0.f, 0.f, 0.f, 0.f, 0.f, 0.f};
#pragma unroll
    for (int ks = 0; ks < 8; ++ks)               // K = 256
      ci = wmma_bf16(load_a(xt, l16, hf, ks, 256), load_bT(Wihb, jt, l16, hf, ks, 256), ci);
    ai[g] = ci;
    v8f ch = {0.f, 0.f, 0.f, 0.f, 0.f, 0.f, 0.f, 0.f};
#pragma unroll
    for (int ks = 0; ks < 4; ++ks)               // K = 128
      ch = wmma_bf16(load_a(ht, l16, hf, ks, 128), load_bT(Whhb, jt, l16, hf, ks, 128), ch);
    ah[g] = ch;
  }

  const int d = wave * 16 + l16;
  const float bi_r = bih[d],       bh_r = bhh[d];
  const float bi_z = bih[128 + d], bh_z = bhh[128 + d];
  const float bi_g = bih[256 + d], bh_g = bhh[256 + d];
#pragma unroll
  for (int v = 0; v < 8; ++v) {
    const int gr = row0 + v + 8 * hf;
    const int grc = gr > N - 1 ? N - 1 : gr;
    const float hv = h[(size_t)grc * 128 + d];   // fp32 state for accuracy
    const float rg = sigmoidf_(ai[0][v] + bi_r + ah[0][v] + bh_r);
    const float zg = sigmoidf_(ai[1][v] + bi_z + ah[1][v] + bh_z);
    const float ng = tanhf_(ai[2][v] + bi_g + rg * (ah[2][v] + bh_g));
    if (gr < N) hout[(size_t)gr * 128 + d] = (1.f - zg) * ng + zg * hv;
  }
}

// ---------------- host orchestration ----------------

extern "C" void kernel_launch(void* const* d_in, const int* in_sizes, int n_in,
                              void* d_out, int out_size, void* d_ws, size_t ws_size,
                              hipStream_t stream) {
  (void)n_in; (void)out_size; (void)ws_size;
  const int EP = in_sizes[4];
  const int V  = in_sizes[6] / DIM;
  const int C  = in_sizes[7] / DIM;

  const int* v_edge = (const int*)d_in[2];
  const int* c_edge = (const int*)d_in[3];
  const int* p_edge = (const int*)d_in[4];
  const int* n_edge = (const int*)d_in[5];
  const float* v_emb = (const float*)d_in[6];
  const float* c_emb = (const float*)d_in[7];

  const float *mW1[4], *mb1[4], *mW2[4], *mb2[4];
  for (int m = 0; m < 4; ++m) {
    mW1[m] = (const float*)d_in[8 + 4 * m + 0];
    mb1[m] = (const float*)d_in[8 + 4 * m + 1];
    mW2[m] = (const float*)d_in[8 + 4 * m + 2];
    mb2[m] = (const float*)d_in[8 + 4 * m + 3];
  }
  const float* cWih = (const float*)d_in[24]; const float* cWhh = (const float*)d_in[25];
  const float* cbih = (const float*)d_in[26]; const float* cbhh = (const float*)d_in[27];
  const float* vWih = (const float*)d_in[28]; const float* vWhh = (const float*)d_in[29];
  const float* vbih = (const float*)d_in[30]; const float* vbhh = (const float*)d_in[31];

  char* ws = (char*)d_ws;
  size_t off = 0;
  auto alloc = [&](size_t bytes) -> void* {
    void* p = ws + off;
    off = (off + bytes + 255) & ~(size_t)255;
    return p;
  };

  u16 *bW1[4], *bW2[4];
  for (int m = 0; m < 4; ++m) {
    bW1[m] = (u16*)alloc(DIM * DIM * sizeof(u16));
    bW2[m] = (u16*)alloc(DIM * DIM * sizeof(u16));
  }
  u16* bcWih = (u16*)alloc(384 * 256 * sizeof(u16));
  u16* bcWhh = (u16*)alloc(384 * 128 * sizeof(u16));
  u16* bvWih = (u16*)alloc(384 * 256 * sizeof(u16));
  u16* bvWhh = (u16*)alloc(384 * 128 * sizeof(u16));

  const size_t maxN = (size_t)(V > C ? V : C);
  float* M     = (float*)alloc(maxN * DIM * sizeof(float));
  float* aggrA = (float*)alloc((size_t)C * DIM * sizeof(float));  // p_v2c_aggr
  float* aggrB = (float*)alloc((size_t)C * DIM * sizeof(float));  // n_v2c_aggr
  float* aggrC = (float*)alloc((size_t)V * DIM * sizeof(float));  // p_c2v_aggr
  float* aggrD = (float*)alloc((size_t)V * DIM * sizeof(float));  // n_c2v_aggr

  auto cast = [&](const float* s, u16* d, int n) {
    k_cast<<<(n + 255) / 256, 256, 0, stream>>>(s, d, n);
  };
  for (int m = 0; m < 4; ++m) {
    cast(mW1[m], bW1[m], DIM * DIM);
    cast(mW2[m], bW2[m], DIM * DIM);
  }
  cast(cWih, bcWih, 384 * 256); cast(cWhh, bcWhh, 384 * 128);
  cast(vWih, bvWih, 384 * 256); cast(vWhh, bvWhh, 384 * 128);

  // d_out layout: stack(v_embs)[3,V,128] then stack(c_embs)[3,C,128]
  float* out = (float*)d_out;
  auto vslot = [&](int i) { return out + (size_t)i * V * DIM; };
  auto cslot = [&](int i) { return out + (size_t)3 * V * DIM + (size_t)i * C * DIM; };

  {
    int n4 = V * (DIM / 4);
    k_copy4<<<(n4 + 255) / 256, 256, 0, stream>>>((const float4*)v_emb, (float4*)vslot(0), n4);
    n4 = C * (DIM / 4);
    k_copy4<<<(n4 + 255) / 256, 256, 0, stream>>>((const float4*)c_emb, (float4*)cslot(0), n4);
  }

  const int scatterBlocks = (EP * 32 + 255) / 256;
  for (int it = 0; it < 2; ++it) {
    const float* vcur = vslot(it);
    const float* ccur = cslot(it);

    // zero all four aggregate buffers (contiguous in ws)
    const int nz4 = (2 * C + 2 * V) * (DIM / 4);
    k_zero4<<<(nz4 + 255) / 256, 256, 0, stream>>>((float4*)aggrA, nz4);

    // p_v2c: MLP over variables, scatter into clause buckets
    k_mlp<<<(V + 127) / 128, 256, 0, stream>>>(vcur, bW1[0], mb1[0], bW2[0], mb2[0], M, V);
    k_scatter<<<scatterBlocks, 256, 0, stream>>>(M, p_edge, v_edge, c_edge, aggrA, EP);
    // n_v2c
    k_mlp<<<(V + 127) / 128, 256, 0, stream>>>(vcur, bW1[1], mb1[1], bW2[1], mb2[1], M, V);
    k_scatter<<<scatterBlocks, 256, 0, stream>>>(M, n_edge, v_edge, c_edge, aggrB, EP);
    // p_c2v: MLP over clauses (pre-update c_emb), scatter into variable buckets
    k_mlp<<<(C + 127) / 128, 256, 0, stream>>>(ccur, bW1[2], mb1[2], bW2[2], mb2[2], M, C);
    k_scatter<<<scatterBlocks, 256, 0, stream>>>(M, p_edge, c_edge, v_edge, aggrC, EP);
    // n_c2v
    k_mlp<<<(C + 127) / 128, 256, 0, stream>>>(ccur, bW1[3], mb1[3], bW2[3], mb2[3], M, C);
    k_scatter<<<scatterBlocks, 256, 0, stream>>>(M, n_edge, c_edge, v_edge, aggrD, EP);

    // clause GRU then variable GRU
    k_gru<<<(C + 15) / 16, 256, 0, stream>>>(aggrA, aggrB, ccur, bcWih, bcWhh, cbih, cbhh, cslot(it + 1), C);
    k_gru<<<(V + 15) / 16, 256, 0, stream>>>(aggrC, aggrD, vcur, bvWih, bvWhh, vbih, vbhh, vslot(it + 1), V);
  }
}
